// ModulatedConv2d_5403068858599
// MI455X (gfx1250) — matrix operand order repaired and checked by reading, not compiled, not run on hardware
//
#include <hip/hip_runtime.h>

// ---------------------------------------------------------------------------
// ModulatedConv2d (StyleGAN2), B=16, CIN=COUT=512, K=3, H=W=64, style=512
// Compute-bound (AI ~1.1 kFLOP/B) -> bf16 WMMA implicit-GEMM conv,
// TDM (tensor_load_to_lds) double-buffered staging, s_wait_tensorcnt sync.
// ---------------------------------------------------------------------------

typedef __attribute__((ext_vector_type(16))) __bf16 v16bf;
typedef __attribute__((ext_vector_type(8)))  float  v8f;
typedef __attribute__((ext_vector_type(4)))  unsigned int u32x4;
typedef __attribute__((ext_vector_type(8)))  unsigned int u32x8;

static constexpr float WG  = 0.014731391274719738f;  // 1/sqrt(512*9)
static constexpr float FCG = 0.044194173824159216f;  // 1/sqrt(512)

__device__ __forceinline__ unsigned short f2bf(float x) {
  unsigned int u = __float_as_uint(x);
  unsigned int r = (u + 0x7fffu + ((u >> 16) & 1u)) >> 16;  // RNE
  return (unsigned short)r;
}

// ---------------------------- prologue kernels -----------------------------

__global__ void affine_kernel(const float* __restrict__ style,
                              const float* __restrict__ fc_w,
                              const float* __restrict__ fc_b,
                              float* __restrict__ s) {
  int idx = blockIdx.x * blockDim.x + threadIdx.x;  // B*CIN = 8192
  int b = idx >> 9, ci = idx & 511;
  const float* st = style + b * 512;
  const float* wr = fc_w + ci * 512;
  float acc = 0.f;
  for (int d = 0; d < 512; ++d) acc += st[d] * wr[d];
  s[idx] = acc * FCG + fc_b[ci];
}

__global__ void wsq_kernel(const float* __restrict__ weight,
                           float* __restrict__ wsq) {
  int idx = blockIdx.x * blockDim.x + threadIdx.x;  // COUT*CIN = 262144
  const float* w = weight + idx * 9;
  float acc = 0.f;
#pragma unroll
  for (int t = 0; t < 9; ++t) acc += w[t] * w[t];
  wsq[idx] = acc;
}

__global__ void deco_kernel(const float* __restrict__ wsq,
                            const float* __restrict__ s,
                            float* __restrict__ deco) {
  int idx = blockIdx.x * blockDim.x + threadIdx.x;  // B*COUT = 8192
  int b = idx >> 9, co = idx & 511;
  const float* sv = s + b * 512;
  const float* wq = wsq + co * 512;
  float acc = 0.f;
  for (int ci = 0; ci < 512; ++ci) { float t = sv[ci]; acc += wq[ci] * t * t; }
  deco[idx] = rsqrtf(acc * WG * WG + 1e-8f);
}

__global__ void wmod_kernel(const float* __restrict__ weight,
                            const float* __restrict__ s,
                            const float* __restrict__ deco,
                            unsigned short* __restrict__ wmod) {
  long long idx = (long long)blockIdx.x * blockDim.x + threadIdx.x;  // 37748736
  int ci = (int)(idx & 511);
  int co = (int)((idx >> 9) & 511);
  int t  = (int)((idx >> 18) % 9);
  int b  = (int)(idx / (512LL * 512LL * 9LL));
  float v = weight[(co * 512 + ci) * 9 + t] * WG * s[b * 512 + ci] * deco[b * 512 + co];
  wmod[idx] = f2bf(v);
}

__global__ void xcast_kernel(const float* __restrict__ x,
                             unsigned short* __restrict__ xb, long long n) {
  long long idx = (long long)blockIdx.x * blockDim.x + threadIdx.x;
  if (idx < n) xb[idx] = f2bf(x[idx]);
}

// ------------------------- TDM descriptor + issue --------------------------
// 3-D tile load per CDNA5 ISA:
//   addr = global_addr + data_size*(x + y*stride_y + z*stride_z)
//   LDS filled contiguously in x-fastest, then y, then z order.
// dims/tiles/strides in 2-byte (bf16) element units. OOB reads return zero.
__device__ __forceinline__ void tdm_load_bf16_3d(
    unsigned lds_off, unsigned long long gaddr,
    unsigned td0, unsigned td1, unsigned td2,        // tensor extents (x,y,z) from tile start
    unsigned t0, unsigned t1, unsigned t2,           // tile dims (x,y,z)
    unsigned long long sy, unsigned long long sz) {  // strides for y and z steps
  u32x4 g0;
  g0[0] = 1u;                                        // count=1, user mode
  g0[1] = lds_off;                                   // lds_addr (bytes)
  g0[2] = (unsigned)gaddr;                           // global_addr[31:0]
  g0[3] = (unsigned)((gaddr >> 32) & 0x01ffffffu) | (2u << 30);  // addr[56:32] | type=2
  u32x8 g1;
  g1[0] = 0x00010000u;                               // data_size=1 (2 bytes), no flags
  g1[1] = (td0 & 0xffffu) << 16;                     // tensor_dim0[15:0]
  g1[2] = (td0 >> 16) | ((td1 & 0xffffu) << 16);     // tensor_dim0[31:16] | tensor_dim1[15:0]
  g1[3] = (td1 >> 16) | (t0 << 16);                  // tensor_dim1[31:16] | tile_dim0
  g1[4] = t1 | (t2 << 16);                           // tile_dim1 | tile_dim2
  g1[5] = (unsigned)sy;                              // tensor_dim0_stride[31:0]
  g1[6] = (unsigned)(sy >> 32) | ((unsigned)(sz & 0xffffu) << 16);
  g1[7] = (unsigned)((sz >> 16) & 0xffffffffu);      // tensor_dim1_stride[47:16]
  u32x4 g2;
  g2[0] = td2;                                       // tensor_dim2 (z extent)
  g2[1] = 0u;
  g2[2] = 0u;
  g2[3] = 0u;
  u32x4 g3 = {0u, 0u, 0u, 0u};
  asm volatile("tensor_load_to_lds %0, %1, %2, %3"
               :: "s"(g0), "s"(g1), "s"(g2), "s"(g3)
               : "memory");
}

// ---------------------------------------------------------------------------
// Main conv: block -> out[b, co0:co0+128, h0:h0+2, 0:64] (128x128 GEMM tile)
// K loop: 16 chunks of 32 input channels x 9 taps, bf16 WMMA 16x16x32.
// LDS (per buffer): As [9][128 co][32 ci] (72KB), xs [4 h][32 ci][64 w] (16KB)
// Double-buffered: 176 KB of the 320 KB WGP LDS.
// ---------------------------------------------------------------------------
union FragA { v16bf v; unsigned int u[8]; unsigned short h[16]; };
union FragB { v16bf v; unsigned int u[8]; unsigned short h[16]; };

#define AS_HALF  36864   // 9*128*32
#define XS_HALF  8192    // 4*32*64

__global__ __launch_bounds__(256) void modconv_wmma_kernel(
    const unsigned short* __restrict__ xb,     // [B][CIN][H][W] bf16
    const unsigned short* __restrict__ wmod,   // [B][9][COUT][CIN] bf16
    float* __restrict__ out) {                 // [B][COUT][H][W] f32
  extern __shared__ unsigned short smem[];
  // halfword layout: As0 | As1 | xs0 | xs1  (no pointer arrays: keep all
  // LDS addressing as plain arithmetic so nothing becomes a static initializer)

  const int tid = threadIdx.x;
  const int co0 = blockIdx.x * 128;  // 4 co tiles
  const int h0  = blockIdx.y * 2;    // 32 row-pairs
  const int b   = blockIdx.z;        // 16 samples

  const int wave = tid >> 5, lane = tid & 31;
  const int wm = wave & 3;   // M group: co rows [wm*32, wm*32+32)
  const int wn = wave >> 2;  // N group: output row h0+wn, cols 0..63
  const int khalf = (lane & 16) ? 8 : 0;
  const int mrow  = lane & 15;

  // LDS byte offset of dynamic shared base (flat LDS address low 32 bits)
  const unsigned smem_base = (unsigned)(unsigned long long)(const void*)smem;

  // x tile geometry (vertical halo; top edge clamped, bottom via TDM OOB zero)
  const int hstart = (h0 == 0) ? 0 : h0 - 1;
  const unsigned nrows = (h0 == 0) ? 3u : 4u;
  const unsigned xls   = (h0 == 0) ? 4096u : 0u;  // skip zeroed slab r=0

  const unsigned long long wbase =
      (unsigned long long)wmod + 2ull * ((unsigned long long)b * 9 * 512 * 512 + (unsigned)co0 * 512);
  const unsigned long long xbase =
      (unsigned long long)xb + 2ull * ((((unsigned long long)b * 512) * 64 + (unsigned)hstart) * 64);

  if (wave == 0) {
    if (h0 == 0) {  // zero slab r=0 (image row -1) of both x buffers, once
      unsigned int* p0 = (unsigned int*)(smem + 4 * AS_HALF);
      unsigned int* p1 = (unsigned int*)(smem + 4 * AS_HALF + XS_HALF);
      for (int i = lane; i < 1024; i += 32) { p0[i] = 0u; p1[i] = 0u; }
    }
    // chunk 0 into buffer 0
    tdm_load_bf16_3d(smem_base,
                     wbase,
                     512u, 512u - (unsigned)co0, 9u, 32u, 128u, 9u, 512ull, 262144ull);
    tdm_load_bf16_3d(smem_base + 8u * AS_HALF + xls,
                     xbase,
                     64u, 512u, 64u - (unsigned)hstart, 64u, 32u, nrows, 4096ull, 64ull);
    __builtin_amdgcn_s_wait_tensorcnt(0);
  }
  __syncthreads();

  v8f zero = {0.f, 0.f, 0.f, 0.f, 0.f, 0.f, 0.f, 0.f};
  v8f acc[2][4];
#pragma unroll
  for (int i = 0; i < 2; ++i)
#pragma unroll
    for (int j = 0; j < 4; ++j) acc[i][j] = zero;

  int buf = 0;
  for (int ci0 = 0; ci0 < 512; ci0 += 32) {
    // prefetch next chunk into other buffer via TDM (overlaps WMMA below)
    if (wave == 0 && ci0 + 32 < 512) {
      const int nb = buf ^ 1;
      const unsigned ci1 = (unsigned)(ci0 + 32);
      tdm_load_bf16_3d(smem_base + (unsigned)nb * (2u * AS_HALF),
                       wbase + 2ull * ci1,
                       512u - ci1, 512u - (unsigned)co0, 9u, 32u, 128u, 9u, 512ull, 262144ull);
      tdm_load_bf16_3d(smem_base + 8u * AS_HALF + (unsigned)nb * (2u * XS_HALF) + xls,
                       xbase + 2ull * ci1 * 4096ull,
                       64u, 512u - ci1, 64u - (unsigned)hstart, 64u, 32u, nrows, 4096ull, 64ull);
    }

    const unsigned short* As = smem + buf * AS_HALF;
    const unsigned short* xs = smem + 4 * AS_HALF + buf * XS_HALF;

    for (int dy3 = 0; dy3 < 3; ++dy3) {  // taps grouped by vertical offset
      // A frags for the 3 taps of this dy (ISA 7.12.2 16-bit A layout)
      FragA a[3][2];
#pragma unroll
      for (int dx3 = 0; dx3 < 3; ++dx3) {
        int t = dy3 * 3 + dx3;
#pragma unroll
        for (int mi = 0; mi < 2; ++mi) {
          int row = wm * 32 + mi * 16 + mrow;
#pragma unroll
          for (int v = 0; v < 8; ++v) {
            int kk = (v < 4 ? 2 * v : 2 * v + 8) + khalf;
            a[dx3][mi].u[v] = *(const unsigned int*)(As + (t * 128 + row) * 32 + kk);
          }
        }
      }
      // This wave's input row for this dy: slab r = wn + dy + 1 = wn + dy3
      const unsigned short* rowp = xs + (dy3 + wn) * 32 * 64;
#pragma unroll
      for (int j = 0; j < 4; ++j) {
        const int base = j * 16;
        const int col = base + (lane & 15);
        FragB fm, f0, fp;  // dx = -1, 0, +1 : shifted views of mid
#pragma unroll
        for (int e = 0; e < 16; ++e) {
          const unsigned short* cp = rowp + ((e < 8 ? e : e + 8) + khalf) * 64;
          f0.h[e] = cp[col];
          fm.h[e] = col ? cp[col - 1] : (unsigned short)0;
          fp.h[e] = (col < 63) ? cp[col + 1] : (unsigned short)0;
        }
#pragma unroll
        for (int mi = 0; mi < 2; ++mi) {
          acc[mi][j] = __builtin_amdgcn_wmma_f32_16x16x32_bf16(
              false, a[0][mi].v, false, fm.v, (short)0, acc[mi][j], false, false);
          acc[mi][j] = __builtin_amdgcn_wmma_f32_16x16x32_bf16(
              false, a[1][mi].v, false, f0.v, (short)0, acc[mi][j], false, false);
          acc[mi][j] = __builtin_amdgcn_wmma_f32_16x16x32_bf16(
              false, a[2][mi].v, false, fp.v, (short)0, acc[mi][j], false, false);
        }
      }
    }

    if (wave == 0) __builtin_amdgcn_s_wait_tensorcnt(0);  // next chunk landed
    __syncthreads();
    buf ^= 1;
  }

  // Epilogue: C/D layout VGPR v, lanes0-15 -> M=v, lanes16-31 -> M=8+v; N=lane&15
  const int oh = h0 + wn;
#pragma unroll
  for (int mi = 0; mi < 2; ++mi)
#pragma unroll
    for (int j = 0; j < 4; ++j)
#pragma unroll
      for (int v = 0; v < 8; ++v) {
        int row = co0 + wm * 32 + mi * 16 + v + ((lane & 16) ? 8 : 0);
        int ow  = j * 16 + (lane & 15);
        out[((((long long)b * 512 + row) << 6) + oh) * 64 + ow] = acc[mi][j][v];
      }
}

// ---------------------------------------------------------------------------
extern "C" void kernel_launch(void* const* d_in, const int* in_sizes, int n_in,
                              void* d_out, int out_size, void* d_ws, size_t ws_size,
                              hipStream_t stream) {
  const float* x      = (const float*)d_in[0];  // [16,512,64,64]
  const float* style  = (const float*)d_in[1];  // [16,512]
  const float* weight = (const float*)d_in[2];  // [1,512,512,3,3]
  const float* fc_w   = (const float*)d_in[3];  // [512,512]
  const float* fc_b   = (const float*)d_in[4];  // [512]
  float* out = (float*)d_out;

  char* ws = (char*)d_ws;
  float* s    = (float*)(ws + 0);                       // 32 KB
  float* wsq  = (float*)(ws + 32768);                   // 1 MB
  float* deco = (float*)(ws + 32768 + 1048576);         // 32 KB
  unsigned short* wmod = (unsigned short*)(ws + 1114112);    // 75.5 MB (L2-resident)
  unsigned short* xbf  = (unsigned short*)(ws + 76611584);   // 67 MB

  affine_kernel<<<32, 256, 0, stream>>>(style, fc_w, fc_b, s);
  wsq_kernel<<<1024, 256, 0, stream>>>(weight, wsq);
  deco_kernel<<<32, 256, 0, stream>>>(wsq, s, deco);
  wmod_kernel<<<147456, 256, 0, stream>>>(weight, s, deco, wmod);
  xcast_kernel<<<131072, 256, 0, stream>>>(x, xbf, 33554432LL);

  dim3 grid(4, 32, 16);  // co tiles x row-pairs x batch
  modconv_wmma_kernel<<<grid, 256, 180224, stream>>>(xbf, wmod, out);
}
